// MultiLatentAttention_27101243638028
// MI455X (gfx1250) — matrix operand order repaired and checked by reading, not compiled, not run on hardware
//
#include <hip/hip_runtime.h>
#include <hip/hip_bf16.h>
#include <stdint.h>

// MI455X / gfx1250, wave32, WMMA bf16 path + async global->LDS staging.
// Assumes ws_size >= ~480MB (five 67MB bf16 tensors + 134MB f32 scratch + weights).

typedef __attribute__((ext_vector_type(16))) __bf16 v16bf;
typedef __attribute__((ext_vector_type(8)))  __bf16 v8bf;
typedef __attribute__((ext_vector_type(8)))  float  v8f;

#define N_B 8
#define N_T 4096
#define N_C 1024
#define N_H 16
#define N_D 64
#define N_L 64
#define N_CS 128
#define N_NC 32
#define N_M (N_B*N_T)

__device__ __forceinline__ unsigned short f2bf_bits(float f){
  union { float f; unsigned u; } x; x.f = f;
  unsigned r = x.u + 0x7FFFu + ((x.u >> 16) & 1u);
  return (unsigned short)(r >> 16);
}
__device__ __forceinline__ __bf16 f2bf(float f){
  union { unsigned short s; __bf16 b; } y; y.s = f2bf_bits(f); return y.b;
}
__device__ __forceinline__ float bf2f(__bf16 b){
  union { unsigned short s; __bf16 b; } y; y.b = b;
  union { unsigned u; float f; } x; x.u = ((unsigned)y.s) << 16;
  return x.f;
}
__device__ __forceinline__ v16bf pack16(v8bf lo, v8bf hi){
  v16bf r;
  #pragma unroll
  for (int i = 0; i < 8; i++){ r[i] = lo[i]; r[i+8] = hi[i]; }
  return r;
}

// ---------------- elementwise converts ----------------
__global__ void k_f32_to_bf16(const float* __restrict__ src, __bf16* __restrict__ dst, int n){
  for (int i = blockIdx.x*blockDim.x + threadIdx.x; i < n; i += gridDim.x*blockDim.x)
    dst[i] = f2bf(src[i]);
}

// latent_queries (1,L,H,D) -> qb (H,L,D) bf16
__global__ void k_q_pack(const float* __restrict__ lat, __bf16* __restrict__ qb){
  int i = blockIdx.x*blockDim.x + threadIdx.x;
  if (i >= N_H*N_L*N_D) return;
  int d = i & 63; int l = (i >> 6) & 63; int h = i >> 12;
  qb[i] = f2bf(lat[(l*N_H + h)*N_D + d]);
}

// raw f32 (B,T,C) k-logits -> RoPE -> kb bf16 (B,H,T,D)
__global__ void k_rope_relayout(const float* __restrict__ raw, __bf16* __restrict__ kb){
  int i = blockIdx.x*blockDim.x + threadIdx.x;      // B*T*H*32 pairs = 2^24
  if (i >= N_B*N_T*N_H*32) return;
  int p = i & 31; int h = (i >> 5) & 15; int t = (i >> 9) & 4095; int b = i >> 21;
  const float* src = raw + ((size_t)(b*N_T + t))*N_C + h*N_D + 2*p;
  float a = src[0], bb = src[1];
  float inv = __expf(-(float)(2*p) * 0.14391156831212787f);  // ln(10000)/64
  float sn, cs; __sincosf((float)t * inv, &sn, &cs);
  unsigned lo = f2bf_bits(a*cs - bb*sn);
  unsigned hi = f2bf_bits(a*sn + bb*cs);
  unsigned* dst = (unsigned*)(kb + ((size_t)((b*N_H + h)*N_T + t))*N_D + 2*p);
  *dst = lo | (hi << 16);
}

// raw f32 (B,T,C) v -> vb bf16 (B,H,T,D)
__global__ void k_v_relayout(const float* __restrict__ raw, __bf16* __restrict__ vb){
  int i = blockIdx.x*blockDim.x + threadIdx.x;      // 2^25
  if (i >= N_M*N_C) return;
  int d = i & 63; int h = (i >> 6) & 15; int t = (i >> 10) & 4095; int b = i >> 22;
  vb[((size_t)((b*N_H + h)*N_T + t))*N_D + d] = f2bf(raw[i]);
}

// raw f32 (B,T,H,L) gate logits -> softmax over L -> gates bf16 (B,H,T,L)
__global__ __launch_bounds__(256) void k_gate_softmax(const float* __restrict__ raw,
                                                      __bf16* __restrict__ gates){
  int wid = threadIdx.x >> 5, lane = threadIdx.x & 31;
  int row = blockIdx.x*8 + wid;                     // (b*T+t)*H + h
  if (row >= N_B*N_T*N_H) return;
  int h = row & 15; int t = (row >> 4) & 4095; int b = row >> 16;
  const float* src = raw + ((size_t)(b*N_T + t))*N_C + h*N_L;
  float v0 = src[lane] * 0.125f;
  float v1 = src[lane+32] * 0.125f;
  float m = fmaxf(v0, v1);
  #pragma unroll
  for (int off = 16; off >= 1; off >>= 1) m = fmaxf(m, __shfl_xor(m, off));
  float e0 = __expf(v0 - m), e1 = __expf(v1 - m);
  float s = e0 + e1;
  #pragma unroll
  for (int off = 16; off >= 1; off >>= 1) s += __shfl_xor(s, off);
  float inv = 1.f/s;
  __bf16* dst = gates + ((size_t)((b*N_H + h)*N_T + t))*N_L;
  dst[lane]    = f2bf(e0*inv);
  dst[lane+32] = f2bf(e1*inv);
}

// ---------------- WMMA bf16 GEMM: out = A(MxK) * B(NxK)^T ----------------
// 128x128 workgroup tile, 8 waves (2x4), per-wave 64x32 = 4x2 WMMA frags,
// K stepped by 32, LDS double buffering.
// GUARD=0: exact-multiple sizes; global->LDS staging via async tensor path
//          (global_load_async_to_lds_b128, ASYNCcnt), no bounds checks.
// GUARD=1: reg-staged path with zero-fill bounds checks (used for N=64 GEMM).
template <int OUT_BF16, int GUARD>
__global__ __launch_bounds__(256) void k_gemm(
    const __bf16* __restrict__ A, int lda, long long aBatch,
    const __bf16* __restrict__ Bm, int ldb, long long bBatch,
    void* __restrict__ outP, int ldo, long long oOuter, long long oInner,
    int M, int N, int K)
{
  __shared__ __align__(16) __bf16 As[2][128*40];
  __shared__ __align__(16) __bf16 Bs[2][128*40];
  const int tid  = threadIdx.x;
  const int lane = tid & 31;
  const int wm   = ((tid >> 5) & 1) * 64;
  const int wn   = (tid >> 6) * 32;
  const int mBase = blockIdx.y * 128;
  const int nBase = blockIdx.x * 128;
  const int z = blockIdx.z;
  const __bf16* Ab = A  + (long long)z * aBatch;
  const __bf16* Bb = Bm + (long long)z * bBatch;
  const long long oOff = (long long)(z >> 4) * oOuter + (long long)(z & 15) * oInner;

  v8f acc[4][2];
  #pragma unroll
  for (int i = 0; i < 4; i++)
    #pragma unroll
    for (int j = 0; j < 2; j++){ v8f zz = {}; acc[i][j] = zz; }

  const int seg = tid & 3;          // 16B segment within 64B row of the k-slab
  const int r0  = tid >> 2;         // tile row (+64 for second slot)
  const int nk  = K >> 5;

  auto computeBlock = [&](int buf){
    // A frag: lane(0-15)=row m, K in {koffA..+7} U {koffA+16..+23}
    const int koffA = (lane >> 4) * 8;
    const int mrow  = wm + (lane & 15);
    v16bf af[4];
    #pragma unroll
    for (int mi = 0; mi < 4; mi++){
      const __bf16* p = &As[buf][(mrow + mi*16)*40 + koffA];
      af[mi] = pack16(*(const v8bf*)p, *(const v8bf*)(p + 16));
    }
    // B frag: lane(0-15)=col n, K in {koffB..+7} U {koffB+16..+23}
    const int koffB = (lane >> 4) * 8;
    const int nrow  = wn + (lane & 15);
    v16bf bfm[2];
    #pragma unroll
    for (int ni = 0; ni < 2; ni++){
      const __bf16* p = &Bs[buf][(nrow + ni*16)*40 + koffB];
      bfm[ni] = pack16(*(const v8bf*)p, *(const v8bf*)(p + 16));
    }
    #pragma unroll
    for (int mi = 0; mi < 4; mi++)
      #pragma unroll
      for (int ni = 0; ni < 2; ni++)
        acc[mi][ni] = __builtin_amdgcn_wmma_f32_16x16x32_bf16(
            false, af[mi], false, bfm[ni], (short)0, acc[mi][ni], false, false);
  };

  if constexpr (!GUARD) {
    // Async global->LDS pipeline (no VGPR staging, no ds_store).
    auto issue = [&](int kb, int buf){
      #pragma unroll
      for (int s = 0; s < 2; s++){
        int row = r0 + s*64;
        unsigned ldsA = (unsigned)(uintptr_t)(const void*)&As[buf][row*40 + seg*8];
        unsigned long long ga =
            (unsigned long long)(uintptr_t)(const void*)(Ab + (long long)(mBase + row)*lda + kb + seg*8);
        asm volatile("global_load_async_to_lds_b128 %0, %1, off"
                     :: "v"(ldsA), "v"(ga) : "memory");
        unsigned ldsB = (unsigned)(uintptr_t)(const void*)&Bs[buf][row*40 + seg*8];
        unsigned long long gb =
            (unsigned long long)(uintptr_t)(const void*)(Bb + (long long)(nBase + row)*ldb + kb + seg*8);
        asm volatile("global_load_async_to_lds_b128 %0, %1, off"
                     :: "v"(ldsB), "v"(gb) : "memory");
      }
    };
    issue(0, 0);
    int buf = 0;
    for (int i = 0; i < nk; i++){
      asm volatile("s_wait_asynccnt 0x0" ::: "memory");  // this wave's LDS fills landed
      __syncthreads();                                   // everyone's landed; prior reads of buf done
      if (i + 1 < nk) issue((i + 1) << 5, buf ^ 1);
      computeBlock(buf);
      buf ^= 1;
    }
  } else {
    // Reg-staged path with zero-fill for partial N tiles.
    const v8bf zero8 = {};
    v8bf ra[2], rb[2];
    auto loadRegs = [&](int kb){
      #pragma unroll
      for (int s = 0; s < 2; s++){
        int row = r0 + s*64;
        int gm = mBase + row;
        ra[s] = (gm < M) ? *(const v8bf*)(Ab + (long long)gm*lda + kb + seg*8) : zero8;
        int gn = nBase + row;
        rb[s] = (gn < N) ? *(const v8bf*)(Bb + (long long)gn*ldb + kb + seg*8) : zero8;
      }
    };
    loadRegs(0);
    int buf = 0;
    for (int i = 0; i < nk; i++){
      #pragma unroll
      for (int s = 0; s < 2; s++){
        *(v8bf*)&As[buf][(r0 + s*64)*40 + seg*8] = ra[s];
        *(v8bf*)&Bs[buf][(r0 + s*64)*40 + seg*8] = rb[s];
      }
      __syncthreads();
      if (i + 1 < nk) loadRegs((i + 1) << 5);
      computeBlock(buf);
      buf ^= 1;
    }
  }

  // C/D layout: vgpr r -> M=r (lanes 0-15) / M=r+8 (lanes 16-31), N=lane%16
  const int nCol = nBase + wn + (lane & 15);
  const int mTop = mBase + wm + ((lane >> 4) << 3);
  #pragma unroll
  for (int mi = 0; mi < 4; mi++)
    #pragma unroll
    for (int ni = 0; ni < 2; ni++){
      int n = nCol + ni*16;
      #pragma unroll
      for (int r = 0; r < 8; r++){
        int m = mTop + mi*16 + r;
        if (GUARD && (m >= M || n >= N)) continue;
        long long idx = oOff + (long long)m*ldo + n;
        if (OUT_BF16) ((__bf16*)outP)[idx] = f2bf(acc[mi][ni][r]);
        else          ((float*)outP)[idx]  = acc[mi][ni][r];
      }
    }
}

// ---------------- latent attention core: one block per (b,h) ----------------
// scores(L=64 x cs=128) = q * Kc^T * 0.125, causal c<=l, softmax over c,
// attn += probs * Vc accumulated over all 32 chunks; emits attn^T bf16 (D,L).
__global__ __launch_bounds__(256) void k_latent_attn(const __bf16* __restrict__ qb,
    const __bf16* __restrict__ kb, const __bf16* __restrict__ vb,
    __bf16* __restrict__ attnT){
  __shared__ __bf16 qs[N_L*N_D];
  __shared__ __bf16 ks[N_CS*N_D];
  __shared__ __bf16 vs[N_CS*N_D];
  __shared__ float accS[N_L*N_D];
  const int tid = threadIdx.x;
  const int bh = blockIdx.x;
  const int h = bh & 15;
  {
    const unsigned* src = (const unsigned*)(qb + (size_t)h*N_L*N_D);
    unsigned* d = (unsigned*)qs;
    for (int i = tid; i < N_L*N_D/2; i += 256) d[i] = src[i];
  }
  for (int i = tid; i < N_L*N_D; i += 256) accS[i] = 0.f;
  const int l = tid >> 2, s = tid & 3;   // 4 lanes per latent row
  const size_t kvbase = (size_t)bh * N_T * N_D;
  for (int n = 0; n < N_NC; n++){
    __syncthreads();
    const unsigned* ksrc = (const unsigned*)(kb + kvbase + (size_t)n*N_CS*N_D);
    const unsigned* vsrc = (const unsigned*)(vb + kvbase + (size_t)n*N_CS*N_D);
    unsigned* kd = (unsigned*)ks; unsigned* vd = (unsigned*)vs;
    for (int i = tid; i < N_CS*N_D/2; i += 256){ kd[i] = ksrc[i]; vd[i] = vsrc[i]; }
    __syncthreads();

    float sc[32];
    float mloc = -3.0e38f;
    #pragma unroll
    for (int j = 0; j < 32; j++){
      int c = s*32 + j;
      float dot = 0.f;
      #pragma unroll 8
      for (int d0 = 0; d0 < N_D; d0++)
        dot = fmaf(bf2f(qs[l*N_D + d0]), bf2f(ks[c*N_D + d0]), dot);
      sc[j] = (c <= l) ? dot * 0.125f : -3.0e38f;
      mloc = fmaxf(mloc, sc[j]);
    }
    float m = mloc;
    m = fmaxf(m, __shfl_xor(m, 1));
    m = fmaxf(m, __shfl_xor(m, 2));
    float pe[32];
    float ssum = 0.f;
    #pragma unroll
    for (int j = 0; j < 32; j++){ float e = __expf(sc[j] - m); pe[j] = e; ssum += e; }
    ssum += __shfl_xor(ssum, 1);
    ssum += __shfl_xor(ssum, 2);
    float inv = 1.f/ssum;
    for (int d0 = 0; d0 < N_D; d0++){
      float part = 0.f;
      #pragma unroll
      for (int j = 0; j < 32; j++)
        part = fmaf(pe[j], bf2f(vs[(s*32 + j)*N_D + d0]), part);
      part += __shfl_xor(part, 1);
      part += __shfl_xor(part, 2);
      if (s == 0) accS[l*N_D + d0] += part * inv;
    }
  }
  __syncthreads();
  for (int i = tid; i < N_L*N_D; i += 256){
    int d0 = i >> 6, l0 = i & 63;
    attnT[(size_t)bh*N_D*N_L + i] = f2bf(accS[l0*N_D + d0]);
  }
}

extern "C" void kernel_launch(void* const* d_in, const int* in_sizes, int n_in,
                              void* d_out, int out_size, void* d_ws, size_t ws_size,
                              hipStream_t stream){
  (void)in_sizes; (void)n_in; (void)out_size; (void)ws_size;
  const float* x   = (const float*)d_in[0];
  const float* lat = (const float*)d_in[1];
  const float* Wk  = (const float*)d_in[2];
  const float* Wv  = (const float*)d_in[3];
  const float* Wg  = (const float*)d_in[4];
  const float* Wp  = (const float*)d_in[5];
  float* out = (float*)d_out;

  char* ws = (char*)d_ws;
  size_t off = 0;
  auto alloc = [&](size_t bytes)->char*{
    char* p = ws + off; off += (bytes + 255) & ~(size_t)255; return p;
  };
  __bf16* xb     = (__bf16*)alloc((size_t)N_M*N_C*2);
  __bf16* wkb    = (__bf16*)alloc((size_t)N_C*N_C*2);
  __bf16* wvb    = (__bf16*)alloc((size_t)N_C*N_C*2);
  __bf16* wgb    = (__bf16*)alloc((size_t)N_C*N_C*2);
  __bf16* wpb    = (__bf16*)alloc((size_t)N_C*N_C*2);
  __bf16* qb     = (__bf16*)alloc((size_t)N_H*N_L*N_D*2);
  __bf16* kbuf   = (__bf16*)alloc((size_t)N_B*N_H*N_T*N_D*2);
  __bf16* vbuf   = (__bf16*)alloc((size_t)N_B*N_H*N_T*N_D*2);
  __bf16* gatesb = (__bf16*)alloc((size_t)N_B*N_H*N_T*N_L*2);
  __bf16* wtd    = (__bf16*)alloc((size_t)N_M*N_C*2);
  __bf16* attnT  = (__bf16*)alloc((size_t)N_B*N_H*N_D*N_L*2);
  float*  raw    = (float*)alloc((size_t)N_M*N_C*4);

  k_f32_to_bf16<<<2048, 256, 0, stream>>>(x,  xb,  N_M*N_C);
  k_f32_to_bf16<<<512,  256, 0, stream>>>(Wk, wkb, N_C*N_C);
  k_f32_to_bf16<<<512,  256, 0, stream>>>(Wv, wvb, N_C*N_C);
  k_f32_to_bf16<<<512,  256, 0, stream>>>(Wg, wgb, N_C*N_C);
  k_f32_to_bf16<<<512,  256, 0, stream>>>(Wp, wpb, N_C*N_C);
  k_q_pack<<<(N_H*N_L*N_D + 255)/256, 256, 0, stream>>>(lat, qb);

  dim3 gBig(N_C/128, N_M/128, 1);  // 8 x 256
  // K logits -> RoPE relayout
  k_gemm<0,0><<<gBig, 256, 0, stream>>>(xb, N_C, 0, wkb, N_C, 0, raw, N_C, 0, 0, N_M, N_C, N_C);
  k_rope_relayout<<<(N_B*N_T*N_H*32)/256, 256, 0, stream>>>(raw, kbuf);
  // V logits -> relayout
  k_gemm<0,0><<<gBig, 256, 0, stream>>>(xb, N_C, 0, wvb, N_C, 0, raw, N_C, 0, 0, N_M, N_C, N_C);
  k_v_relayout<<<(N_M*N_C)/256, 256, 0, stream>>>(raw, vbuf);
  // Gate logits -> softmax
  k_gemm<0,0><<<gBig, 256, 0, stream>>>(xb, N_C, 0, wgb, N_C, 0, raw, N_C, 0, 0, N_M, N_C, N_C);
  k_gate_softmax<<<(N_B*N_T*N_H)/8, 256, 0, stream>>>(raw, gatesb);
  // Latent attention core -> attn^T (B,H,D,L) bf16
  k_latent_attn<<<N_B*N_H, 256, 0, stream>>>(qb, kbuf, vbuf, attnT);
  // weighted(b,t,h*64+d) = gates_bh(T x L) * attn_bh(L x D), batched over z=b*16+h
  dim3 gW(1, N_T/128, N_B*N_H);
  k_gemm<1,1><<<gW, 256, 0, stream>>>(gatesb, N_L, (long long)N_T*N_L,
                                      attnT,  N_L, (long long)N_D*N_L,
                                      wtd, N_C, (long long)N_T*N_C, 64,
                                      N_T, N_D, N_L);
  // final projection -> d_out f32
  k_gemm<0,0><<<gBig, 256, 0, stream>>>(wtd, N_C, 0, wpb, N_C, 0, out, N_C, 0, 0, N_M, N_C, N_C);
}
